// SSMMemoryModule_60559038873727
// MI455X (gfx1250) — compile-verified, matrix operand order checked by reading, not run on hardware
//
#include <hip/hip_runtime.h>
#include <hip/hip_bf16.h>
#include <math.h>

// ---------------------------------------------------------------------------
// SSM memory module for MI455X (gfx1250, wave32).
// HBM-bound (~340MB @ 23.3TB/s => ~15us floor, AI ~38 FLOP/B). All GEMMs on
// v_wmma_f32_16x16x32_bf16 (fp32 accumulate). bf16 conversion minimized:
// reused operands pre-converted to bf16 in workspace (raw uint loads into the
// WMMA register image); streamed operands converted once via v_perm_b32
// packing. One wave per block, 4 accumulators per wave (16x64 tile) so every
// large stream is read exactly once per GEMM.
// ---------------------------------------------------------------------------

#define DEV __device__ __forceinline__

constexpr int BATCH = 4;
constexpr int DS    = 64;
constexpr int DM    = 4096;
constexpr int NE    = 1024;
constexpr int SLEN  = 2048;

typedef __attribute__((ext_vector_type(16))) __bf16        v16bf;
typedef __attribute__((ext_vector_type(8)))  float         v8f;
typedef __attribute__((ext_vector_type(8)))  unsigned int  v8u;

// ---- helpers --------------------------------------------------------------

// pack two f32 into two bf16 (round-half-up) in one v_perm_b32
DEV unsigned int pkbf(float x, float y) {
    const unsigned int ux = __builtin_bit_cast(unsigned int, x) + 0x8000u;
    const unsigned int uy = __builtin_bit_cast(unsigned int, y) + 0x8000u;
    return __builtin_amdgcn_perm(uy, ux, 0x07060302);   // {uy.hi16, ux.hi16}
}
DEV unsigned short bf16s(float x) {
    return (unsigned short)((__builtin_bit_cast(unsigned int, x) + 0x8000u) >> 16);
}

DEV v8f wmma_bf16(v8u a, v8u b, v8f c) {
    return __builtin_amdgcn_wmma_f32_16x16x32_bf16(
        false, __builtin_bit_cast(v16bf, a),
        false, __builtin_bit_cast(v16bf, b),
        (short)0, c, false, false);
}

DEV float waveReduceSum(float v) {
    #pragma unroll
    for (int o = 16; o > 0; o >>= 1) v += __shfl_xor(v, o, 32);
    return v;
}
DEV float waveReduceMax(float v) {
    #pragma unroll
    for (int o = 16; o > 0; o >>= 1) v = fmaxf(v, __shfl_xor(v, o, 32));
    return v;
}

// byte-pair K offset (p = dword 0..7) inside a 16x16x32 bf16 WMMA operand for
// this lane-half (hi = lane>>4). Pair (k,k+1) is contiguous in row-major mem.
DEV int kpair(int p, int hi) {
    return ((p < 4) ? (2 * p) : (16 + 2 * (p - 4))) + 8 * hi;
}

// ---- 0: bulk f32 -> bf16 (for W_key / W_query) ----------------------------
__global__ __launch_bounds__(256)
void cvt_bf16_kernel(const float* __restrict__ src, unsigned int* __restrict__ dst,
                     int npairs) {
    const int i = blockIdx.x * 256 + threadIdx.x;
    if (i < npairs) {
        const float2 v = ((const float2*)src)[i];
        dst[i] = pkbf(v.x, v.y);
    }
}

// ---- 1 & 5: projection  out = X[b] (MxDM) @ Wbf^T (DMx64) -----------------
// One wave per block; wave computes a 16(m) x 64(n) tile with 4 accumulators,
// loading each A element exactly once. B is pre-converted bf16 (raw loads).
// KEYS=1: fp32 out, layout [B][64][M] (feeds softmax / update A-operand)
// KEYS=0: bf16 out, layout [B][M][64] (feeds readout A-operand)
template <bool KEYS>
__global__ __launch_bounds__(32)
void proj_kernel(const float* __restrict__ X, const unsigned short* __restrict__ Wbf,
                 void* __restrict__ outp, int M) {
    const int mb   = blockIdx.x % (M / 16);
    const int b    = blockIdx.x / (M / 16);
    const int lane = threadIdx.x;
    const int hi   = lane >> 4;
    const int lm   = lane & 15;
    const int m0   = mb * 16;

    const float* __restrict__ arow = X + ((size_t)b * M + m0 + lm) * DM;
    const unsigned short* __restrict__ wrow[4];
    #pragma unroll
    for (int j = 0; j < 4; ++j) wrow[j] = Wbf + (size_t)(j * 16 + lm) * DM;

    v8f acc[4] = {};
    for (int kb = 0; kb < DM; kb += 32) {
        v8u a;
        #pragma unroll
        for (int p = 0; p < 8; ++p) {
            const int k = kb + kpair(p, hi);
            const float2 av = *(const float2*)(arow + k);
            a[p] = pkbf(av.x, av.y);
        }
        #pragma unroll
        for (int j = 0; j < 4; ++j) {
            v8u bm;
            #pragma unroll
            for (int p = 0; p < 8; ++p)
                bm[p] = *(const unsigned int*)(wrow[j] + kb + kpair(p, hi));
            acc[j] = wmma_bf16(a, bm, acc[j]);      // 4 independent WMMAs
        }
    }

    const int mhalf = 8 * hi;                       // C/D: row m = v + 8*hi
    if (KEYS) {
        float* o = (float*)outp;
        #pragma unroll
        for (int j = 0; j < 4; ++j) {
            float* oj = o + ((size_t)b * DS + j * 16 + lm) * M + m0 + mhalf;
            #pragma unroll
            for (int v = 0; v < 8; ++v) oj[v] = acc[j][v];   // 32B contiguous
        }
    } else {
        unsigned short* o = (unsigned short*)outp;
        #pragma unroll
        for (int j = 0; j < 4; ++j)
            #pragma unroll
            for (int v = 0; v < 8; ++v)
                o[((size_t)b * M + m0 + mhalf + v) * DS + j * 16 + lm] = bf16s(acc[j][v]);
    }
}

// ---- 2: softmax over NE; fp32 in (keysT), bf16 out (wwbf) -----------------
__global__ __launch_bounds__(128)
void softmax_col_kernel(const float* __restrict__ kT, unsigned short* __restrict__ wwbf) {
    const float* p = kT + (size_t)blockIdx.x * NE;      // one (b,s) row of 1024
    unsigned short* w = wwbf + (size_t)blockIdx.x * NE;
    const int t = threadIdx.x;
    float vals[8];
    float mx = -INFINITY;
    #pragma unroll
    for (int j = 0; j < 8; ++j) { vals[j] = p[t + 128 * j]; mx = fmaxf(mx, vals[j]); }

    __shared__ float redM[4];
    __shared__ float redS[4];
    mx = waveReduceMax(mx);
    if ((t & 31) == 0) redM[t >> 5] = mx;
    __syncthreads();
    mx = fmaxf(fmaxf(redM[0], redM[1]), fmaxf(redM[2], redM[3]));

    float sum = 0.f;
    #pragma unroll
    for (int j = 0; j < 8; ++j) { vals[j] = expf(vals[j] - mx); sum += vals[j]; }
    sum = waveReduceSum(sum);
    if ((t & 31) == 0) redS[t >> 5] = sum;
    __syncthreads();
    sum = redS[0] + redS[1] + redS[2] + redS[3];

    const float inv = 1.0f / sum;
    #pragma unroll
    for (int j = 0; j < 8; ++j) w[t + 128 * j] = bf16s(vals[j] * inv);
}

// ---- 2b: summary = mean_n E[b,n,:] ----------------------------------------
__global__ __launch_bounds__(256)
void summary_kernel(const float* __restrict__ E, float* __restrict__ summary) {
    const int chunks = DM / 256;
    const int b = blockIdx.x / chunks;
    const int d = (blockIdx.x % chunks) * 256 + threadIdx.x;
    const float* p = E + (size_t)b * NE * DM + d;
    float s = 0.f;
    for (int n = 0; n < NE; ++n) s += p[(size_t)n * DM];
    summary[b * DM + d] = s * (1.0f / NE);
}

// ---- 2c: beta = sigmoid(Wb @ summary + bb) --------------------------------
__global__ __launch_bounds__(128)
void beta_kernel(const float* __restrict__ Wb, const float* __restrict__ bb,
                 const float* __restrict__ summary, float* __restrict__ beta) {
    const int s = blockIdx.x % DS;
    const int b = blockIdx.x / DS;
    const float* w  = Wb + (size_t)s * DM;
    const float* sm = summary + (size_t)b * DM;
    float acc = 0.f;
    for (int d = threadIdx.x; d < DM; d += 128) acc += w[d] * sm[d];
    acc = waveReduceSum(acc);
    __shared__ float red[4];
    if ((threadIdx.x & 31) == 0) red[threadIdx.x >> 5] = acc;
    __syncthreads();
    if (threadIdx.x == 0) {
        const float t = red[0] + red[1] + red[2] + red[3] + bb[s];
        beta[b * DS + s] = 1.0f / (1.0f + expf(-t));
    }
}

// ---- 3: new_state^T (bf16) = (beta*state + ww^T @ E)^T --------------------
// One wave per block: 64(m=DS, 4 accs) x 16(n) tile => evicted stream read
// exactly once. A (wwbf) is bf16 raw loads. Out layout [B][DM][DS] bf16.
__global__ __launch_bounds__(32)
void update_state_kernel(const unsigned short* __restrict__ wwbf, const float* __restrict__ E,
                         const float* __restrict__ beta, const float* __restrict__ state,
                         unsigned short* __restrict__ nsT) {
    const int nb   = blockIdx.x % (DM / 16);
    const int b    = blockIdx.x / (DM / 16);
    const int lane = threadIdx.x;
    const int hi   = lane >> 4;
    const int lm   = lane & 15;
    const int n    = nb * 16 + lm;

    const unsigned short* __restrict__ arow[4];
    #pragma unroll
    for (int j = 0; j < 4; ++j) arow[j] = wwbf + ((size_t)b * DS + j * 16 + lm) * NE;
    const float* __restrict__ Eb = E + (size_t)b * NE * DM;

    v8f acc[4] = {};
    for (int kb = 0; kb < NE; kb += 32) {
        v8u bm;
        #pragma unroll
        for (int p = 0; p < 8; ++p) {
            const int k = kb + kpair(p, hi);
            bm[p] = pkbf(Eb[(size_t)k * DM + n], Eb[(size_t)(k + 1) * DM + n]);
        }
        #pragma unroll
        for (int j = 0; j < 4; ++j) {
            v8u a;
            #pragma unroll
            for (int p = 0; p < 8; ++p)
                a[p] = *(const unsigned int*)(arow[j] + kb + kpair(p, hi));
            acc[j] = wmma_bf16(a, bm, acc[j]);
        }
    }

    const int mhalf = 8 * hi;
    #pragma unroll
    for (int j = 0; j < 4; ++j) {
        float vals[8];
        #pragma unroll
        for (int v = 0; v < 8; ++v) {
            const int ms = j * 16 + mhalf + v;
            vals[v] = beta[b * DS + ms] * state[((size_t)b * DS + ms) * DM + n] + acc[j][v];
        }
        unsigned int* o = (unsigned int*)(nsT + ((size_t)b * DM + n) * DS + j * 16 + mhalf);
        #pragma unroll
        for (int v = 0; v < 4; ++v) o[v] = pkbf(vals[2 * v], vals[2 * v + 1]); // 16B
    }
}

// ---- 4b: gate = sigmoid(X @ gw^T + gb), one token per wave ----------------
__global__ __launch_bounds__(256)
void gate_kernel(const float* __restrict__ X, const float* __restrict__ gw,
                 const float* __restrict__ gb, float* __restrict__ gate) {
    const int tok  = blockIdx.x * 8 + (threadIdx.x >> 5);   // flat over B*S
    const int lane = threadIdx.x & 31;
    const float* x = X + (size_t)tok * DM;
    float acc = 0.f;
    for (int d = lane; d < DM; d += 32) acc += x[d] * gw[d];
    acc = waveReduceSum(acc);
    if (lane == 0) gate[tok] = 1.0f / (1.0f + expf(-(acc + gb[0])));
}

// ---- 6: out = gate ⊙ (q @ new_state); all operands pre-converted bf16 -----
// One wave per block: 16(m) x 64(n, 4 accs); K=64 => 2x4 WMMAs; zero cvt VALU.
__global__ __launch_bounds__(32)
void readout_kernel(const unsigned short* __restrict__ qbf, const unsigned short* __restrict__ nsT,
                    const float* __restrict__ gate, float* __restrict__ out) {
    const int NBN = DM / 64;
    const int NBM = SLEN / 16;
    const int nb = blockIdx.x % NBN;
    const int mb = (blockIdx.x / NBN) % NBM;
    const int b  = blockIdx.x / (NBN * NBM);
    const int lane = threadIdx.x;
    const int hi   = lane >> 4;
    const int lm   = lane & 15;
    const int m0   = mb * 16;

    const unsigned short* __restrict__ qrow = qbf + ((size_t)b * SLEN + m0 + lm) * DS;
    const unsigned short* __restrict__ nrow[4];
    #pragma unroll
    for (int j = 0; j < 4; ++j)
        nrow[j] = nsT + ((size_t)b * DM + nb * 64 + j * 16 + lm) * DS;

    v8f acc[4] = {};
    #pragma unroll
    for (int kb = 0; kb < DS; kb += 32) {
        v8u a;
        #pragma unroll
        for (int p = 0; p < 8; ++p)
            a[p] = *(const unsigned int*)(qrow + kb + kpair(p, hi));
        #pragma unroll
        for (int j = 0; j < 4; ++j) {
            v8u bm;
            #pragma unroll
            for (int p = 0; p < 8; ++p)
                bm[p] = *(const unsigned int*)(nrow[j] + kb + kpair(p, hi));
            acc[j] = wmma_bf16(a, bm, acc[j]);
        }
    }

    const int mhalf = 8 * hi;
    float g[8];
    #pragma unroll
    for (int v = 0; v < 8; ++v) g[v] = gate[b * SLEN + m0 + mhalf + v];
    #pragma unroll
    for (int j = 0; j < 4; ++j)
        #pragma unroll
        for (int v = 0; v < 8; ++v) {
            const int tok = m0 + mhalf + v;
            out[((size_t)b * SLEN + tok) * DM + nb * 64 + j * 16 + lm] = g[v] * acc[j][v];
        }
}

// ---------------------------------------------------------------------------

extern "C" void kernel_launch(void* const* d_in, const int* in_sizes, int n_in,
                              void* d_out, int out_size, void* d_ws, size_t ws_size,
                              hipStream_t stream) {
    (void)in_sizes; (void)n_in; (void)out_size; (void)ws_size;

    const float* state   = (const float*)d_in[0];   // [B,DS,DM]
    const float* evicted = (const float*)d_in[1];   // [B,NE,DM]
    const float* current = (const float*)d_in[2];   // [B,S,DM]
    const float* W_key   = (const float*)d_in[3];   // [DS,DM]
    const float* W_bw    = (const float*)d_in[4];   // [DS,DM]
    const float* W_bb    = (const float*)d_in[5];   // [DS]
    const float* W_query = (const float*)d_in[6];   // [DS,DM]
    const float* gate_w  = (const float*)d_in[7];   // [1,DM]
    const float* gate_b  = (const float*)d_in[8];   // [1]
    float* out = (float*)d_out;                     // [B,S,DM]

    // workspace carve-up (256B aligned)
    char* ws = (char*)d_ws;
    size_t off = 0;
    auto carve = [&](size_t bytes) { char* p = ws + off; off = (off + bytes + 255) & ~(size_t)255; return p; };
    unsigned short* wkbf    = (unsigned short*)carve((size_t)DS * DM * 2);           // bf16 W_key
    unsigned short* wqbf    = (unsigned short*)carve((size_t)DS * DM * 2);           // bf16 W_query
    float*          keysT   = (float*)carve((size_t)BATCH * DS * NE * 4);            // [B][DS][NE]
    unsigned short* wwbf    = (unsigned short*)carve((size_t)BATCH * DS * NE * 2);   // bf16 weights
    float*          summary = (float*)carve((size_t)BATCH * DM * 4);
    float*          beta    = (float*)carve((size_t)BATCH * DS * 4);
    unsigned short* qbf     = (unsigned short*)carve((size_t)BATCH * SLEN * DS * 2); // [B][S][DS]
    float*          gatews  = (float*)carve((size_t)BATCH * SLEN * 4);
    unsigned short* nsT     = (unsigned short*)carve((size_t)BATCH * DM * DS * 2);   // [B][DM][DS]

    const int wpairs = DS * DM / 2;
    cvt_bf16_kernel<<<(wpairs + 255) / 256, 256, 0, stream>>>(W_key,   (unsigned int*)wkbf, wpairs);
    cvt_bf16_kernel<<<(wpairs + 255) / 256, 256, 0, stream>>>(W_query, (unsigned int*)wqbf, wpairs);

    // compress phase
    proj_kernel<true><<<BATCH * (NE / 16), 32, 0, stream>>>(evicted, wkbf, keysT, NE);
    softmax_col_kernel<<<BATCH * DS, 128, 0, stream>>>(keysT, wwbf);
    summary_kernel<<<BATCH * (DM / 256), 256, 0, stream>>>(evicted, summary);
    beta_kernel<<<BATCH * DS, 128, 0, stream>>>(W_bw, W_bb, summary, beta);
    update_state_kernel<<<BATCH * (DM / 16), 32, 0, stream>>>(wwbf, evicted, beta, state, nsT);

    // query phase
    proj_kernel<false><<<BATCH * (SLEN / 16), 32, 0, stream>>>(current, wqbf, qbf, SLEN);
    gate_kernel<<<(BATCH * SLEN) / 8, 256, 0, stream>>>(current, gate_w, gate_b, gatews);
    readout_kernel<<<BATCH * (SLEN / 16) * (DM / 64), 32, 0, stream>>>(qbf, nsT, gatews, out);
}
